// SelfAttention_90529320665967
// MI455X (gfx1250) — matrix operand order repaired and checked by reading, not compiled, not run on hardware
//
#include <hip/hip_runtime.h>
#include <hip/hip_bf16.h>

#define BB  4
#define CC  128
#define HWD 4096   // 64*64

typedef __attribute__((ext_vector_type(8)))  float  v8f;
typedef __attribute__((ext_vector_type(16))) __bf16 v16bf;
typedef __attribute__((ext_vector_type(8)))  __bf16 v8bf;

__device__ __forceinline__ v16bf cat8(v8bf lo, v8bf hi) {
  return __builtin_shufflevector(lo, hi, 0,1,2,3,4,5,6,7,8,9,10,11,12,13,14,15);
}

__device__ __forceinline__ v8f wmma_bf16(v16bf a, v16bf b, v8f c) {
  return __builtin_amdgcn_wmma_f32_16x16x32_bf16(false, a, false, b, (short)0, c,
                                                 false, false);
}

// ---------------------------------------------------------------------------
// Kernel 1: Q/K/V projections (1x1 conv == GEMM M=HW, N=C, K=C).
// Q,K stored row-major (hw, c) bf16; V stored transposed (c, hw) bf16.
// ---------------------------------------------------------------------------
__global__ __launch_bounds__(256)
void sa_proj_kernel(const float* __restrict__ X,
                    const float* __restrict__ Wq, const float* __restrict__ bq,
                    const float* __restrict__ Wk, const float* __restrict__ bk,
                    const float* __restrict__ Wv, const float* __restrict__ bv,
                    __bf16* __restrict__ q, __bf16* __restrict__ k,
                    __bf16* __restrict__ vt)
{
  const int wave  = (int)(blockIdx.x * blockDim.x + threadIdx.x) >> 5;
  const int lane  = (int)threadIdx.x & 31;
  const int b     = wave >> 8;            // 256 i-tiles per batch
  const int it    = wave & 255;
  const int ibase = it << 4;
  const int lm    = lane & 15;
  const int kgA   = (lane < 16) ? 0 : 8;  // A-operand / D-row group
  const int kgB   = (lane < 16) ? 0 : 16; // B-operand K group

  const float* Xb = X + (size_t)b * CC * HWD;
  const int iA = ibase + lm;              // this lane's A-matrix row (pixel)

  for (int mtx = 0; mtx < 3; ++mtx) {
    const float* Wm = (mtx == 0) ? Wq : (mtx == 1) ? Wk : Wv;
    const float* bm = (mtx == 0) ? bq : (mtx == 1) ? bk : bv;

    v8f acc[8];
    #pragma unroll
    for (int n = 0; n < 8; ++n) acc[n] = (v8f)0.0f;

    for (int kc = 0; kc < 4; ++kc) {
      // A[i, c] = X[b, c, i]  (stride-HW gathers, converted to bf16)
      v16bf a;
      #pragma unroll
      for (int e = 0; e < 16; ++e) {
        const int c = kc * 32 + kgB + e;
        a[e] = (__bf16)Xb[(size_t)c * HWD + iA];
      }
      #pragma unroll
      for (int n = 0; n < 8; ++n) {
        // B[c, o] = W[o, c]: per-lane contiguous 16 floats of row o
        const float* wrow = Wm + (size_t)(n * 16 + lm) * CC + kc * 32 + kgB;
        v16bf bw;
        #pragma unroll
        for (int e = 0; e < 16; ++e) bw[e] = (__bf16)wrow[e];
        acc[n] = wmma_bf16(a, bw, acc[n]);
      }
    }

    #pragma unroll
    for (int n = 0; n < 8; ++n) {
      const int o = n * 16 + lm;
      const float bias = bm[o];
      #pragma unroll
      for (int r = 0; r < 8; ++r) {
        const int i = ibase + kgA + r;    // D layout: M = r + (lane<16?0:8)
        const float val = acc[n][r] + bias;
        if (mtx == 0)      q[((size_t)b * HWD + i) * CC + o]  = (__bf16)val;
        else if (mtx == 1) k[((size_t)b * HWD + i) * CC + o]  = (__bf16)val;
        else               vt[((size_t)b * CC + o) * HWD + i] = (__bf16)val;
      }
    }
  }
}

// ---------------------------------------------------------------------------
// Kernel 2: per-key-column softmax stats (softmax over the QUERY axis).
// One wave owns a 16-wide j-tile; online max / sum-exp over all i.
// ---------------------------------------------------------------------------
__global__ __launch_bounds__(256)
void sa_stats_kernel(const __bf16* __restrict__ q, const __bf16* __restrict__ k,
                     float* __restrict__ mOut, float* __restrict__ dOut)
{
  const int wave  = (int)(blockIdx.x * blockDim.x + threadIdx.x) >> 5;
  const int lane  = (int)threadIdx.x & 31;
  const int b     = wave >> 8;
  const int jt    = wave & 255;
  const int jbase = jt << 4;
  const int lm    = lane & 15;
  const int kgA   = (lane < 16) ? 0 : 8;
  const int kgB   = (lane < 16) ? 0 : 16;

  const __bf16* Qb   = q + (size_t)b * HWD * CC;
  const __bf16* krow = k + ((size_t)b * HWD + jbase + lm) * CC; // B row K[j,*]

  // B operand (K^T) is loop-invariant for this wave: preload 4 K-chunks
  v16bf bt[4];
  #pragma unroll
  for (int kc = 0; kc < 4; ++kc)
    bt[kc] = cat8(*(const v8bf*)(krow + kc * 32 + kgA), *(const v8bf*)(krow + kc * 32 + kgA + 16));

  float m_run = -3.0e38f;
  float d_run = 0.0f;

  for (int it = 0; it < 256; ++it) {
    const __bf16* qrow = Qb + (size_t)(it * 16 + lm) * CC;
    v8f s = (v8f)0.0f;
    #pragma unroll
    for (int kc = 0; kc < 4; ++kc) {
      v8bf lo = *(const v8bf*)(qrow + kc * 32 + kgA);
      v8bf hi = *(const v8bf*)(qrow + kc * 32 + kgA + 16);
      s = wmma_bf16(cat8(lo, hi), bt[kc], s);
    }
    // column j = jbase + lm; lane L holds rows r+kgA, lane L^16 the other half
    float tmax = s[0];
    #pragma unroll
    for (int r = 1; r < 8; ++r) tmax = fmaxf(tmax, s[r]);
    tmax = fmaxf(tmax, __shfl_xor(tmax, 16, 32));
    const float mn    = fmaxf(m_run, tmax);
    const float scale = __expf(m_run - mn);
    float ls = 0.0f;
    #pragma unroll
    for (int r = 0; r < 8; ++r) ls += __expf(s[r] - mn);
    ls += __shfl_xor(ls, 16, 32);
    d_run = d_run * scale + ls;
    m_run = mn;
  }
  if (lane < 16) {
    mOut[(size_t)b * HWD + jbase + lane] = m_run;
    dOut[(size_t)b * HWD + jbase + lane] = d_run;
  }
}

// ---------------------------------------------------------------------------
// Kernel 3: out = softmax(S)·V + residual. Transposed score tiles T = K·Q^T
// land exactly in the A-operand layout for the P·V WMMA (no shuffles).
// ---------------------------------------------------------------------------
__global__ __launch_bounds__(256)
void sa_out_kernel(const float* __restrict__ X,
                   const __bf16* __restrict__ q, const __bf16* __restrict__ k,
                   const __bf16* __restrict__ vt,
                   const float* __restrict__ mIn, const float* __restrict__ dIn,
                   float* __restrict__ out)
{
  const int wave  = (int)(blockIdx.x * blockDim.x + threadIdx.x) >> 5;
  const int lane  = (int)threadIdx.x & 31;
  const int b     = wave >> 8;
  const int it    = wave & 255;
  const int ibase = it << 4;
  const int lm    = lane & 15;
  const int kgA   = (lane < 16) ? 0 : 8;
  const int kgB   = (lane < 16) ? 0 : 16;

  const __bf16* Qb  = q  + (size_t)b * HWD * CC;
  const __bf16* Kb  = k  + (size_t)b * HWD * CC;
  const __bf16* Vtb = vt + (size_t)b * CC * HWD;
  const float*  mB  = mIn + (size_t)b * HWD;
  const float*  dB  = dIn + (size_t)b * HWD;

  // B operand for T = K·Q^T:  B[c, i] = Q[i, c]; loop-invariant, preload
  const __bf16* qrow = Qb + (size_t)(ibase + lm) * CC;
  v16bf bq4[4];
  #pragma unroll
  for (int kc = 0; kc < 4; ++kc)
    bq4[kc] = cat8(*(const v8bf*)(qrow + kc * 32 + kgA), *(const v8bf*)(qrow + kc * 32 + kgA + 16));

  v8f acc[8];
  #pragma unroll
  for (int n = 0; n < 8; ++n) acc[n] = (v8f)0.0f;

  for (int jc = 0; jc < 128; ++jc) {       // 32 keys per iteration
    const int j0 = jc * 32;
    const __bf16* krow0 = Kb + (size_t)(j0 + lm) * CC;
    const __bf16* krow1 = Kb + (size_t)(j0 + 16 + lm) * CC;
    v8f t0 = (v8f)0.0f, t1 = (v8f)0.0f;
    #pragma unroll
    for (int kc = 0; kc < 4; ++kc) {
      v8bf lo0 = *(const v8bf*)(krow0 + kc * 32 + kgA);
      v8bf hi0 = *(const v8bf*)(krow0 + kc * 32 + kgA + 16);
      t0 = wmma_bf16(cat8(lo0, hi0), bq4[kc], t0);
      v8bf lo1 = *(const v8bf*)(krow1 + kc * 32 + kgA);
      v8bf hi1 = *(const v8bf*)(krow1 + kc * 32 + kgA + 16);
      t1 = wmma_bf16(cat8(lo1, hi1), bq4[kc], t1);
    }
    // T layout: lane column = query i = lm; VGPR r = key j0 + kgA + r.
    // p[j,i] = exp(t - m_j)/d_j; packs directly into the 16x32 A operand.
    v16bf pa;
    const float* mp0 = mB + j0 + kgA;
    const float* dp0 = dB + j0 + kgA;
    #pragma unroll
    for (int r = 0; r < 8; ++r)
      pa[r] = (__bf16)(__expf(t0[r] - mp0[r]) / dp0[r]);
    const float* mp1 = mB + j0 + 16 + kgA;
    const float* dp1 = dB + j0 + 16 + kgA;
    #pragma unroll
    for (int r = 0; r < 8; ++r)
      pa[8 + r] = (__bf16)(__expf(t1[r] - mp1[r]) / dp1[r]);

    #pragma unroll
    for (int n = 0; n < 8; ++n) {
      // B = V (j, c) read from transposed store: two 8-bf16 runs at j0+kgA and j0+16+kgA
      v16bf bvv = cat8(*(const v8bf*)(Vtb + (size_t)(n * 16 + lm) * HWD + j0 + kgA), *(const v8bf*)(Vtb + (size_t)(n * 16 + lm) * HWD + j0 + 16 + kgA));
      acc[n] = wmma_bf16(pa, bvv, acc[n]);
    }
  }

  // Residual epilogue: out[b, c, i] = X[b, c, i] + attn; contiguous per lane
  #pragma unroll
  for (int n = 0; n < 8; ++n) {
    const int c = n * 16 + lm;
    const size_t base = ((size_t)b * CC + c) * HWD + ibase + kgA;
    const float* xp = X + base;
    float* op = out + base;
    #pragma unroll
    for (int r = 0; r < 8; ++r) op[r] = xp[r] + acc[n][r];
  }
}

// ---------------------------------------------------------------------------
extern "C" void kernel_launch(void* const* d_in, const int* in_sizes, int n_in,
                              void* d_out, int out_size, void* d_ws, size_t ws_size,
                              hipStream_t stream) {
  (void)in_sizes; (void)n_in; (void)out_size; (void)ws_size;
  const float* X  = (const float*)d_in[0];
  const float* Wq = (const float*)d_in[1];
  const float* bq = (const float*)d_in[2];
  const float* Wk = (const float*)d_in[3];
  const float* bk = (const float*)d_in[4];
  const float* Wv = (const float*)d_in[5];
  const float* bv = (const float*)d_in[6];
  float* out = (float*)d_out;

  // Workspace: Q (4MB) | K (4MB) | V^T (4MB) | m (64KB) | d (64KB)
  __bf16* qw  = (__bf16*)d_ws;
  __bf16* kw  = qw  + (size_t)BB * HWD * CC;
  __bf16* vtw = kw  + (size_t)BB * HWD * CC;
  float*  mw  = (float*)(vtw + (size_t)BB * HWD * CC);
  float*  dw  = mw + (size_t)BB * HWD;

  dim3 blk(256);                       // 8 wave32 waves per block
  dim3 grd(BB * (HWD / 16) / 8);       // 1024 tile-waves -> 128 blocks

  sa_proj_kernel <<<grd, blk, 0, stream>>>(X, Wq, bq, Wk, bk, Wv, bv, qw, kw, vtw);
  sa_stats_kernel<<<grd, blk, 0, stream>>>(qw, kw, mw, dw);
  sa_out_kernel  <<<grd, blk, 0, stream>>>(X, qw, kw, vtw, mw, dw, out);
}
